// Autoformer_49374944035122
// MI455X (gfx1250) — matrix operand order repaired and checked by reading, not compiled, not run on hardware
//
#include <hip/hip_runtime.h>

// ---------------------------------------------------------------------------
// Autoformer forward for MI455X (gfx1250, wave32).
// All dense GEMMs run on v_wmma_f32_16x16x32_bf16 (fp32 accum).
// Input flattening assumption: setup_inputs() dict traversed in insertion
// order, recursively:
//   0: x [16,768,7]
//   enc_emb: 1 conv_w [512,7,3], 2 conv_b [512]
//   dec_emb: 3 conv_w, 4 conv_b
//   enc_layers[0]: 5 Wq 6 bq 7 Wk 8 bk 9 Wv 10 bv 11 Wo 12 bo 13 w1 14 b1 15 w2 16 b2
//   enc_layers[1]: 17..28 (same pattern)
//   dec_layers[0]: self 29..36, cross 37..44, 45 proj_w [1,512,3], 46 proj_b,
//                  47 w1 48 b1 49 w2 50 b2
//   51: Wp [512,1], 52: bp [1]
// ---------------------------------------------------------------------------

#define B_    16
#define L_    768
#define CIN_  7
#define D_    512
#define F_    2048
#define T_    (B_*L_)
#define KMA_  25
#define TOPK_ 6
#define PRED_ 192

typedef __attribute__((ext_vector_type(16))) __bf16 v16bf;
typedef __attribute__((ext_vector_type(8)))  float  v8f;

// float -> bf16 (round-to-nearest-even), kept as raw ushort for LDS staging
__device__ inline unsigned short f2bf(float f) {
  union { float f; unsigned u; } v; v.f = f;
  unsigned u = v.u + 0x7FFFu + ((v.u >> 16) & 1u);
  return (unsigned short)(u >> 16);
}

__device__ inline float block_reduce_sum_256(float val) {
  __shared__ float red[256];
  int t = threadIdx.x;
  red[t] = val;
  __syncthreads();
  for (int o = 128; o > 0; o >>= 1) {
    if (t < o) red[t] += red[t + o];
    __syncthreads();
  }
  float r = red[0];
  __syncthreads();
  return r;
}

// ---------------------------------------------------------------------------
// WMMA GEMM: C[M,N] = A[M,K] * B + bias (+relu) (+resid)
//   BNK=false : weight stored [K,N] row-major (x @ W convention)
//   BNK=true  : weight stored [N,K] row-major (einsum 'bld,fd->blf' convention)
// Block: 256 threads = 8 waves; block tile 128x64; wave tile 32x32 (2x2 WMMA).
// ---------------------------------------------------------------------------
template <bool BNK>
__global__ __launch_bounds__(256)
void gemm_wmma_kernel(const float* __restrict__ A, const float* __restrict__ Bw,
                      const float* __restrict__ bias, const float* __restrict__ resid,
                      float* __restrict__ C, int M, int N, int K, int relu)
{
  __shared__ unsigned short As[128][34]; // [row][k], +2 pad
  __shared__ unsigned short Bs[64][34];  // [col n][k], +2 pad

  const int tid  = threadIdx.x;
  const int lane = tid & 31;
  const int wave = tid >> 5;
  const int wm0  = (wave >> 1) * 32; // 0,32,64,96
  const int wn0  = (wave & 1) * 32;  // 0,32
  const int p    = lane & 15;
  const int hi   = lane >> 4;
  const int m_base = blockIdx.x * 128;
  const int n_base = blockIdx.y * 64;

  v8f acc[2][2];
#pragma unroll
  for (int i = 0; i < 2; i++)
#pragma unroll
    for (int j = 0; j < 2; j++)
#pragma unroll
      for (int r = 0; r < 8; r++) acc[i][j][r] = 0.0f;

  for (int kb = 0; kb < K; kb += 32) {
    // ---- stage A tile (128x32), fp32 -> bf16 ----
#pragma unroll
    for (int it = 0; it < 4; it++) {
      int idx  = tid + it * 256;     // 1024 = 128 rows * 8 quads
      int row  = idx >> 3;
      int quad = idx & 7;
      const float4 a4 = *(const float4*)(A + (size_t)(m_base + row) * K + kb + quad * 4);
      unsigned short* dst = &As[row][quad * 4];
      dst[0] = f2bf(a4.x); dst[1] = f2bf(a4.y);
      dst[2] = f2bf(a4.z); dst[3] = f2bf(a4.w);
    }
    // ---- stage B tile as [n][k] (64x32) ----
    if (BNK) {
#pragma unroll
      for (int it = 0; it < 2; it++) {
        int idx  = tid + it * 256;   // 512 = 64 n * 8 quads
        int n    = idx >> 3;
        int quad = idx & 7;
        const float4 b4 = *(const float4*)(Bw + (size_t)(n_base + n) * K + kb + quad * 4);
        unsigned short* dst = &Bs[n][quad * 4];
        dst[0] = f2bf(b4.x); dst[1] = f2bf(b4.y);
        dst[2] = f2bf(b4.z); dst[3] = f2bf(b4.w);
      }
    } else {
#pragma unroll
      for (int it = 0; it < 8; it++) {
        int idx = tid + it * 256;    // 2048 = 32 k * 64 n
        int k   = idx >> 6;
        int n   = idx & 63;
        Bs[n][k] = f2bf(Bw[(size_t)(kb + k) * N + n_base + n]);
      }
    }
    __syncthreads();

    // ---- fragments per ISA VGPR layouts ----
    v16bf afrag[2], bfrag[2];
#pragma unroll
    for (int i = 0; i < 2; i++) {
      const unsigned short* ar = &As[wm0 + i * 16 + p][0];
      union { v16bf v; unsigned short s[16]; } u;
#pragma unroll
      for (int e = 0; e < 8; e++) {
        u.s[e]     = ar[hi * 8 + e];        // K = 8*hi + e
        u.s[8 + e] = ar[16 + hi * 8 + e];   // K = 16 + 8*hi + e
      }
      afrag[i] = u.v;
    }
#pragma unroll
    for (int j = 0; j < 2; j++) {
      const unsigned short* br = &Bs[wn0 + j * 16 + p][0];
      union { v16bf v; unsigned short s[16]; } u;
#pragma unroll
      for (int e = 0; e < 8; e++) {
        u.s[e]     = br[hi * 8 + e];        // K = 8*hi + e
        u.s[8 + e] = br[16 + hi * 8 + e];   // K = 16 + 8*hi + e
      }
      bfrag[j] = u.v;
    }

#pragma unroll
    for (int i = 0; i < 2; i++)
#pragma unroll
      for (int j = 0; j < 2; j++)
        acc[i][j] = __builtin_amdgcn_wmma_f32_16x16x32_bf16(
            false, afrag[i], false, bfrag[j], (short)0, acc[i][j], false, false);
    __syncthreads();
  }

  // ---- epilogue: D vgpr r -> M = r + 8*hi, N = p ----
#pragma unroll
  for (int i = 0; i < 2; i++) {
#pragma unroll
    for (int j = 0; j < 2; j++) {
      int n = n_base + wn0 + j * 16 + p;
      float bv = bias ? bias[n] : 0.0f;
#pragma unroll
      for (int r = 0; r < 8; r++) {
        int m = m_base + wm0 + i * 16 + hi * 8 + r;
        float val = acc[i][j][r] + bv;
        if (relu) val = fmaxf(val, 0.0f);
        if (resid) val += resid[(size_t)m * N + n];
        C[(size_t)m * N + n] = val;
      }
    }
  }
}

// ---------------------------------------------------------------------------
// Series decomposition: moving average k=25, zero padded (count_include_pad).
// seasonal = x - mm, trend = mm (either pointer may be null).
// ---------------------------------------------------------------------------
__global__ void decomp_kernel(const float* __restrict__ x, float* __restrict__ seasonal,
                              float* __restrict__ trend, int C)
{
  int i = blockIdx.x * blockDim.x + threadIdx.x;
  int total = B_ * L_ * C;
  if (i >= total) return;
  int c  = i % C;
  int bt = i / C;
  int t  = bt % L_;
  int b  = bt / L_;
  const int pad = (KMA_ - 1) / 2;
  float s = 0.0f;
#pragma unroll 5
  for (int j = -pad; j <= pad; j++) {
    int tj = t + j;
    if (tj >= 0 && tj < L_) s += x[((size_t)b * L_ + tj) * C + c];
  }
  float mm = s * (1.0f / KMA_);
  if (seasonal) seasonal[i] = x[i] - mm;
  if (trend)    trend[i]    = mm;
}

// ---------------------------------------------------------------------------
// Token embedding: circular conv1d (kernel 3, pad 1) + sinusoidal pos-embed.
// ---------------------------------------------------------------------------
__global__ void embed_kernel(const float* __restrict__ x, const float* __restrict__ cw,
                             const float* __restrict__ cb, float* __restrict__ out)
{
  int i = blockIdx.x * blockDim.x + threadIdx.x; // over B*L*D
  if (i >= T_ * D_) return;
  int co = i & (D_ - 1);
  int bt = i / D_;
  int t  = bt % L_;
  int b  = bt / L_;
  float s = cb[co];
#pragma unroll
  for (int j = 0; j < 3; j++) {
    int tj = t + j - 1;
    if (tj < 0) tj += L_;
    if (tj >= L_) tj -= L_;
    const float* xr = x + ((size_t)b * L_ + tj) * CIN_;
    const float* wr = cw + (size_t)co * CIN_ * 3 + j;
#pragma unroll
    for (int ci = 0; ci < CIN_; ci++) s += xr[ci] * wr[ci * 3];
  }
  // positional embedding: pe[t, 2k] = sin(t*exp(-2k*ln(1e4)/D)), pe[t,2k+1]=cos(..)
  const float c0   = -9.210340371976184f / (float)D_; // -ln(10000)/D
  int   base = co & ~1;
  float freq = __expf((float)base * c0);
  float ang  = (float)t * freq;
  float pe   = (co & 1) ? __cosf(ang) : __sinf(ang);
  out[i] = s + pe;
}

// ---------------------------------------------------------------------------
// mean_value[b,tau] = (1/D) * sum_t dot(q[b,t,:], k[b,(t-tau)%L,:])
// (channel-mean of the circular cross-correlation; FFT eliminated)
// One block per (b,tau).
// ---------------------------------------------------------------------------
__global__ __launch_bounds__(256)
void meanvalue_kernel(const float* __restrict__ q, const float* __restrict__ k,
                      float* __restrict__ mv)
{
  int b   = blockIdx.x / L_;
  int tau = blockIdx.x % L_;
  const float4* q4 = (const float4*)(q + (size_t)b * L_ * D_);
  const float4* k4 = (const float4*)(k + (size_t)b * L_ * D_);
  float s = 0.0f;
  for (int i = threadIdx.x; i < L_ * (D_ / 4); i += 256) {
    int t  = i >> 7;       // D_/4 = 128
    int d4 = i & 127;
    int ts = t - tau; if (ts < 0) ts += L_;
    float4 a = q4[t * 128 + d4];
    float4 c = k4[ts * 128 + d4];
    s += a.x * c.x + a.y * c.y + a.z * c.z + a.w * c.w;
  }
  s = block_reduce_sum_256(s);
  if (threadIdx.x == 0) mv[blockIdx.x] = s * (1.0f / D_);
}

// ---------------------------------------------------------------------------
// Global top-6 delays of batch-mean correlation + per-batch softmax weights.
// Single block.
// ---------------------------------------------------------------------------
__global__ void topk_softmax_kernel(const float* __restrict__ mv,
                                    int* __restrict__ idx_out, float* __restrict__ w_out)
{
  __shared__ float g[L_];
  __shared__ int   sidx[TOPK_];
  int tid = threadIdx.x;
  for (int t = tid; t < L_; t += blockDim.x) {
    float s = 0.0f;
    for (int b = 0; b < B_; b++) s += mv[b * L_ + t];
    g[t] = s * (1.0f / B_);
  }
  __syncthreads();
  if (tid == 0) {
    for (int i = 0; i < TOPK_; i++) {
      int bi = 0; float bv = -1e30f;
      for (int t = 0; t < L_; t++) if (g[t] > bv) { bv = g[t]; bi = t; }
      sidx[i] = bi;
      g[bi] = -1e30f;
      idx_out[i] = bi;
    }
  }
  __syncthreads();
  if (tid < B_) {
    float w[TOPK_];
    float mx = -1e30f;
    for (int i = 0; i < TOPK_; i++) { w[i] = mv[tid * L_ + sidx[i]]; mx = fmaxf(mx, w[i]); }
    float s = 0.0f;
    for (int i = 0; i < TOPK_; i++) { w[i] = __expf(w[i] - mx); s += w[i]; }
    float inv = 1.0f / s;
    for (int i = 0; i < TOPK_; i++) w_out[tid * TOPK_ + i] = w[i] * inv;
  }
}

// out[b,t,d] = sum_i w[b,i] * v[b,(t+idx_i)%L,d]   (roll(vals, -idx) blend)
__global__ void agg_kernel(const float* __restrict__ v, const int* __restrict__ idx,
                           const float* __restrict__ w, float* __restrict__ out)
{
  int i = blockIdx.x * blockDim.x + threadIdx.x;
  if (i >= T_ * D_) return;
  int d  = i & (D_ - 1);
  int bt = i / D_;
  int t  = bt % L_;
  int b  = bt / L_;
  float s = 0.0f;
#pragma unroll
  for (int j = 0; j < TOPK_; j++) {
    int tt = t + idx[j];
    if (tt >= L_) tt -= L_;
    s += w[b * TOPK_ + j] * v[((size_t)b * L_ + tt) * D_ + d];
  }
  out[i] = s;
}

__global__ void add3_kernel(const float* a, const float* b, const float* c, float* out, int n)
{
  int i = blockIdx.x * blockDim.x + threadIdx.x;
  if (i < n) out[i] = a[i] + b[i] + c[i];
}

// residual trend projection: circular conv1d over d_model -> 1 channel
__global__ __launch_bounds__(256)
void trendproj_kernel(const float* __restrict__ tsum, const float* __restrict__ pw,
                      const float* __restrict__ pb, float* __restrict__ rt)
{
  int bl = blockIdx.x;           // 0..B*L-1
  int b  = bl / L_;
  int t  = bl % L_;
  int tm = (t == 0) ? (L_ - 1) : (t - 1);
  int tp = (t == L_ - 1) ? 0 : (t + 1);
  const float* r0 = tsum + ((size_t)b * L_ + tm) * D_;
  const float* r1 = tsum + ((size_t)b * L_ + t) * D_;
  const float* r2 = tsum + ((size_t)b * L_ + tp) * D_;
  float s = 0.0f;
  for (int d = threadIdx.x; d < D_; d += 256)
    s += r0[d] * pw[d * 3 + 0] + r1[d] * pw[d * 3 + 1] + r2[d] * pw[d * 3 + 2];
  s = block_reduce_sum_256(s);
  if (threadIdx.x == 0) rt[bl] = s + pb[0];
}

// out[b,p] = sum_d (dec[b,l,d] + rt[b,l]) * Wp[d] + bp,  l = L-192+p
__global__ __launch_bounds__(256)
void final_kernel(const float* __restrict__ dec, const float* __restrict__ rt,
                  const float* __restrict__ Wp, const float* __restrict__ bpp,
                  float* __restrict__ out)
{
  int bi = blockIdx.x;             // 0..B*PRED-1
  int b  = bi / PRED_;
  int p  = bi % PRED_;
  int l  = L_ - PRED_ + p;
  float r = rt[b * L_ + l];
  const float* row = dec + ((size_t)b * L_ + l) * D_;
  float s = 0.0f;
  for (int d = threadIdx.x; d < D_; d += 256) s += (row[d] + r) * Wp[d];
  s = block_reduce_sum_256(s);
  if (threadIdx.x == 0) out[b * PRED_ + p] = s + bpp[0];
}

// ---------------------------------------------------------------------------
// Host-side orchestration
// ---------------------------------------------------------------------------
struct Attn { const float *Wq, *bq, *Wk, *bk, *Wv, *bv, *Wo, *bo; };

static inline Attn attn_at(void* const* in, int i) {
  Attn a;
  a.Wq = (const float*)in[i + 0]; a.bq = (const float*)in[i + 1];
  a.Wk = (const float*)in[i + 2]; a.bk = (const float*)in[i + 3];
  a.Wv = (const float*)in[i + 4]; a.bv = (const float*)in[i + 5];
  a.Wo = (const float*)in[i + 6]; a.bo = (const float*)in[i + 7];
  return a;
}

static void gemm(hipStream_t s, const float* A, const float* Bw, const float* bias,
                 const float* resid, float* C, int M, int N, int K, bool bnk, bool relu)
{
  dim3 grid(M / 128, N / 64), blk(256);
  if (bnk) gemm_wmma_kernel<true ><<<grid, blk, 0, s>>>(A, Bw, bias, resid, C, M, N, K, relu ? 1 : 0);
  else     gemm_wmma_kernel<false><<<grid, blk, 0, s>>>(A, Bw, bias, resid, C, M, N, K, relu ? 1 : 0);
}

static void run_decomp(hipStream_t s, const float* x, float* seasonal, float* trend, int C)
{
  int total = B_ * L_ * C;
  decomp_kernel<<<(total + 255) / 256, 256, 0, s>>>(x, seasonal, trend, C);
}

static void run_attention(hipStream_t s, const float* xq, const float* xkv, const Attn& a,
                          float* q, float* k, float* v, float* agg, float* outy,
                          float* mv, float* wsm, int* idx, const float* resid)
{
  gemm(s, xq,  a.Wq, a.bq, nullptr, q, T_, D_, D_, false, false);
  gemm(s, xkv, a.Wk, a.bk, nullptr, k, T_, D_, D_, false, false);
  gemm(s, xkv, a.Wv, a.bv, nullptr, v, T_, D_, D_, false, false);
  meanvalue_kernel<<<B_ * L_, 256, 0, s>>>(q, k, mv);
  topk_softmax_kernel<<<1, 256, 0, s>>>(mv, idx, wsm);
  agg_kernel<<<(T_ * D_ + 255) / 256, 256, 0, s>>>(v, idx, wsm, agg);
  gemm(s, agg, a.Wo, a.bo, resid, outy, T_, D_, D_, false, false);
}

static void run_ffn(hipStream_t s, const float* x, const float* w1, const float* b1,
                    const float* w2, const float* b2, float* h, float* outy)
{
  gemm(s, x, w1, b1, nullptr, h,    T_, F_, D_, true, true);  // relu(x@w1^T+b1)
  gemm(s, h, w2, b2, x,       outy, T_, D_, F_, true, false); // x + h@w2^T+b2
}

extern "C" void kernel_launch(void* const* d_in, const int* in_sizes, int n_in,
                              void* d_out, int out_size, void* d_ws, size_t ws_size,
                              hipStream_t stream)
{
  (void)in_sizes; (void)n_in; (void)out_size; (void)ws_size;

  const float* x        = (const float*)d_in[0];
  const float* enc_cw   = (const float*)d_in[1];
  const float* enc_cb   = (const float*)d_in[2];
  const float* dec_cw   = (const float*)d_in[3];
  const float* dec_cb   = (const float*)d_in[4];
  // encoder layers
  Attn ea0 = attn_at(d_in, 5);
  const float *e0w1 = (const float*)d_in[13], *e0b1 = (const float*)d_in[14];
  const float *e0w2 = (const float*)d_in[15], *e0b2 = (const float*)d_in[16];
  Attn ea1 = attn_at(d_in, 17);
  const float *e1w1 = (const float*)d_in[25], *e1b1 = (const float*)d_in[26];
  const float *e1w2 = (const float*)d_in[27], *e1b2 = (const float*)d_in[28];
  // decoder layer
  Attn dself  = attn_at(d_in, 29);
  Attn dcross = attn_at(d_in, 37);
  const float *proj_w = (const float*)d_in[45], *proj_b = (const float*)d_in[46];
  const float *dw1 = (const float*)d_in[47], *db1 = (const float*)d_in[48];
  const float *dw2 = (const float*)d_in[49], *db2 = (const float*)d_in[50];
  const float *Wp  = (const float*)d_in[51], *bp  = (const float*)d_in[52];

  float* out = (float*)d_out;

  // ---- workspace carve ----
  const size_t TD = (size_t)T_ * D_;
  float* W    = (float*)d_ws;
  float* enc  = W;             // A0
  float* dec  = enc + TD;      // A1
  float* qb   = dec + TD;      // A2 (qb..ab contiguous: FFN hidden h aliases them)
  float* kb   = qb  + TD;      // A3
  float* vb   = kb  + TD;      // A4
  float* ab   = vb  + TD;      // A5
  float* yb   = ab  + TD;      // A6
  float* x2b  = yb  + TD;      // A7
  float* t1   = x2b + TD;      // A8
  float* t2   = t1  + TD;      // A9
  float* t3   = t2  + TD;      // A10
  float* seas7 = t3 + TD;                          // [B,L,7]
  float* mv    = seas7 + (size_t)B_ * L_ * CIN_;   // [B,L]
  float* wsm   = mv + (size_t)B_ * L_;             // [B,TOPK]
  float* rt    = wsm + (size_t)B_ * TOPK_;         // [B,L]
  int*   idx   = (int*)(rt + (size_t)B_ * L_);     // [TOPK]
  float* h     = qb;                               // [T, 2048] alias of A2..A5

  // ---- embeddings ----
  run_decomp(stream, x, seas7, nullptr, CIN_);                 // seasonal_init
  embed_kernel<<<(T_ * D_ + 255) / 256, 256, 0, stream>>>(x,     enc_cw, enc_cb, enc);
  embed_kernel<<<(T_ * D_ + 255) / 256, 256, 0, stream>>>(seas7, dec_cw, dec_cb, dec);

  // ---- encoder layers ----
  {
    const Attn*  A[2]  = { &ea0, &ea1 };
    const float* W1[2] = { e0w1, e1w1 };
    const float* B1[2] = { e0b1, e1b1 };
    const float* W2[2] = { e0w2, e1w2 };
    const float* B2[2] = { e0b2, e1b2 };
    for (int l = 0; l < 2; l++) {
      run_attention(stream, enc, enc, *A[l], qb, kb, vb, ab, yb, mv, wsm, idx, enc);
      run_decomp(stream, yb, x2b, nullptr, D_);               // x, _ = decomp(x+attn)
      run_ffn(stream, x2b, W1[l], B1[l], W2[l], B2[l], h, yb);
      run_decomp(stream, yb, enc, nullptr, D_);               // res, _ = decomp(x+ffn)
    }
  }

  // ---- decoder layer ----
  run_attention(stream, dec, dec, dself, qb, kb, vb, ab, yb, mv, wsm, idx, dec);
  run_decomp(stream, yb, dec, t1, D_);                        // x, t1 = decomp
  run_attention(stream, dec, enc, dcross, qb, kb, vb, ab, yb, mv, wsm, idx, dec);
  run_decomp(stream, yb, dec, t2, D_);                        // x, t2 = decomp
  run_ffn(stream, dec, dw1, db1, dw2, db2, h, yb);
  run_decomp(stream, yb, dec, t3, D_);                        // x, t3 = decomp

  // residual trend: conv1d_circ(t1+t2+t3) -> [B,L,1]
  add3_kernel<<<((int)TD + 255) / 256, 256, 0, stream>>>(t1, t2, t3, yb, (int)TD);
  trendproj_kernel<<<B_ * L_, 256, 0, stream>>>(yb, proj_w, proj_b, rt);

  // final projection, last PRED_ steps
  final_kernel<<<B_ * PRED_, 256, 0, stream>>>(dec, rt, Wp, bp, out);
}
